// RoPEAttention_85667417686150
// MI455X (gfx1250) — matrix operand order repaired and checked by reading, not compiled, not run on hardware
//
#include <hip/hip_runtime.h>

typedef __attribute__((ext_vector_type(16))) __bf16 v16bf;
typedef __attribute__((ext_vector_type(8)))  __bf16 bf16x8;
typedef __attribute__((ext_vector_type(8)))  float  v8f;
typedef __attribute__((ext_vector_type(4)))  unsigned int u32x4;
typedef __attribute__((ext_vector_type(8)))  int i32x8;
typedef __attribute__((ext_vector_type(4)))  int i32x4;

#define DIM    1024
#define HEADS  16
#define DHEAD  64
#define INNER  1024
#define SEQ    2048
#define BATCH  4
#define ROWS   (BATCH*SEQ)      /* 8192 */
#define SCALE  0.125f           /* 64^-0.5 */

#if __has_builtin(__builtin_amdgcn_tensor_load_to_lds)
#define USE_TDM 1
#else
#define USE_TDM 0
#endif

#if __has_include(<hip/amd_detail/amd_gfx1250_TDM.h>)
#define TDM_SIX_ARG 1
#else
#define TDM_SIX_ARG 0
#endif

__device__ __forceinline__ v8f v8f_zero() {
  v8f z;
#pragma unroll
  for (int i = 0; i < 8; ++i) z[i] = 0.f;
  return z;
}

__device__ __forceinline__ v16bf ld16(const __bf16* p0, const __bf16* p1) {
  union { v16bf v; bf16x8 h[2]; } u;
  u.h[0] = *(const bf16x8*)p0;
  u.h[1] = *(const bf16x8*)p1;
  return u.v;
}

__device__ __forceinline__ v8f wmma_bf16(v16bf a, v16bf b, v8f c) {
  // D = A(16x32 bf16) * B(32x16 bf16) + C(16x16 f32)
  return __builtin_amdgcn_wmma_f32_16x16x32_bf16(false, a, false, b, (short)0, c,
                                                 false, false);
}

#if USE_TDM
// Low 32 bits of a generic pointer into LDS = byte offset in LDS space.
__device__ __forceinline__ unsigned lds_off(const void* p) {
  return (unsigned)(size_t)p;
}

// TDM 2-D tile load: (height x 32) bf16 elements, row stride = stride_elems,
// packed row-major into LDS at lds_byte_off (row pitch 64 B).
__device__ __forceinline__ void tdm_load_tile_2d(unsigned lds_byte_off,
                                                 const void* gaddr,
                                                 int height, int stride_elems) {
  u32x4 g0;
  g0[0] = 1u;                                         // count=1, user mode
  g0[1] = lds_byte_off;                               // lds_addr (bytes)
  unsigned long long ga = (unsigned long long)(size_t)gaddr;
  g0[2] = (unsigned)ga;                               // global_addr[31:0]
  g0[3] = (unsigned)((ga >> 32) & 0x01ffffffull)      // global_addr[56:32]
        | (2u << 30);                                 // type=2 ("image")
  i32x8 g1;
  g1[0] = 1 << 16;                                    // data_size = 2 bytes
  g1[1] = 32 << 16;                                   // tensor_dim0[15:0]=32
  g1[2] = (height & 0xffff) << 16;                    // tensor_dim1[15:0]
  g1[3] = 32 << 16;                                   // tile_dim0 = 32
  g1[4] = height & 0xffff;                            // tile_dim1 (tile_dim2=0)
  g1[5] = stride_elems;                               // tensor_dim0_stride[31:0]
  g1[6] = 0;
  g1[7] = 0;
  i32x4 z4;
#pragma unroll
  for (int i = 0; i < 4; ++i) z4[i] = 0;
#if TDM_SIX_ARG
  i32x8 z8;
#pragma unroll
  for (int i = 0; i < 8; ++i) z8[i] = 0;
  __builtin_amdgcn_tensor_load_to_lds(g0, g1, z4, z4, z8, 0);
#else
  __builtin_amdgcn_tensor_load_to_lds(g0, g1, z4, z4, 0);
#endif
}
#endif  // USE_TDM

// ---------------------------------------------------------------------------
// 1) LayerNorm: x[row][1024] f32 -> xn bf16
// ---------------------------------------------------------------------------
__global__ __launch_bounds__(256) void ln_kernel(const float* __restrict__ x,
                                                 const float* __restrict__ gamma,
                                                 const float* __restrict__ beta,
                                                 __bf16* __restrict__ out) {
  __shared__ float red[256];
  const int row = blockIdx.x;
  const int t = threadIdx.x;
  const float* xr = x + (size_t)row * DIM;
  float v[4];
  float s = 0.f;
#pragma unroll
  for (int i = 0; i < 4; ++i) { v[i] = xr[t + 256 * i]; s += v[i]; }
  red[t] = s; __syncthreads();
  for (int st = 128; st > 0; st >>= 1) { if (t < st) red[t] += red[t + st]; __syncthreads(); }
  const float mu = red[0] * (1.f / DIM);
  __syncthreads();
  float vs = 0.f;
#pragma unroll
  for (int i = 0; i < 4; ++i) { float d = v[i] - mu; vs += d * d; }
  red[t] = vs; __syncthreads();
  for (int st = 128; st > 0; st >>= 1) { if (t < st) red[t] += red[t + st]; __syncthreads(); }
  const float rs = rsqrtf(red[0] * (1.f / DIM) + 1e-5f);
  __bf16* orow = out + (size_t)row * DIM;
#pragma unroll
  for (int i = 0; i < 4; ++i) {
    int c = t + 256 * i;
    orow[c] = (__bf16)((v[i] - mu) * rs * gamma[c] + beta[c]);
  }
}

// ---------------------------------------------------------------------------
// 2) Weight transpose f32[K][N] -> bf16[N][K]  (K-contiguous for B-fragments)
// ---------------------------------------------------------------------------
__global__ __launch_bounds__(256) void transpose_kernel(const float* __restrict__ W,
                                                        __bf16* __restrict__ Wt,
                                                        int K, int N) {
  int idx = blockIdx.x * 256 + threadIdx.x;
  if (idx < K * N) {
    int k = idx / N, n = idx - k * N;
    Wt[(size_t)n * K + k] = (__bf16)W[idx];
  }
}

// ---------------------------------------------------------------------------
// 3) Tiled WMMA GEMM: C[M][N] = A[M][K](bf16) * Bt[N][K]^T (bf16)
//    tile 128x64, Kblk=32; 8 waves, each wave = 16x64 strip (4 C fragments).
//    Operand staging: double-buffered TDM (tensor_load_to_lds + TENSORcnt).
// ---------------------------------------------------------------------------
__global__ __launch_bounds__(256) void gemm_wmma_bf16(const __bf16* __restrict__ A,
                                                      const __bf16* __restrict__ Bt,
                                                      int M, int N, int K,
                                                      __bf16* __restrict__ Cbf,
                                                      float* __restrict__ Cf,
                                                      const float* __restrict__ bias) {
  __shared__ __align__(16) __bf16 sA[2][128 * 32];
  __shared__ __align__(16) __bf16 sB[2][64 * 32];
  const int t = threadIdx.x;
  const int wave = t >> 5, lane = t & 31;
  const int l16 = lane & 15, lhalf = (lane >> 4) & 1;
  const int m0 = blockIdx.x * 128, n0 = blockIdx.y * 64;
  const int nk = K >> 5;

  v8f acc[4];
#pragma unroll
  for (int i = 0; i < 4; ++i) acc[i] = v8f_zero();

#if USE_TDM
  // ---- prologue: DMA first k-block into buffer 0 ------------------------
  if (wave == 0) {
    tdm_load_tile_2d(lds_off(&sA[0][0]), &A[(size_t)m0 * K], 128, K);
    tdm_load_tile_2d(lds_off(&sB[0][0]), &Bt[(size_t)n0 * K], 64, K);
  }
  for (int kb = 0; kb < nk; ++kb) {
    const int cur = kb & 1;
    if (wave == 0) {
      if (kb + 1 < nk) {
        const int nxt = (kb + 1) & 1;
        // issue next tiles, then retire current pair (TDM in-order per wave)
        tdm_load_tile_2d(lds_off(&sA[nxt][0]),
                         &A[(size_t)m0 * K + (size_t)(kb + 1) * 32], 128, K);
        tdm_load_tile_2d(lds_off(&sB[nxt][0]),
                         &Bt[(size_t)n0 * K + (size_t)(kb + 1) * 32], 64, K);
        __builtin_amdgcn_s_wait_tensorcnt((short)2);
      } else {
        __builtin_amdgcn_s_wait_tensorcnt((short)0);
      }
    }
    __syncthreads();
    // A fragment: lane row = l16, K = {lhalf*8..+7} and {16+lhalf*8..+7}
    const __bf16* ap = &sA[cur][(wave * 16 + l16) * 32 + lhalf * 8];
    v16bf afrag = ld16(ap, ap + 16);
#pragma unroll
    for (int nt = 0; nt < 4; ++nt) {
      // B fragment: lane col = l16, K = {lhalf*8..+7} and {16+lhalf*8..+7}
      const __bf16* bp = &sB[cur][(nt * 16 + l16) * 32 + lhalf * 8];
      acc[nt] = wmma_bf16(afrag, ld16(bp, bp + 16), acc[nt]);
    }
    __syncthreads();
  }
#else
  // ---- fallback: synchronous VGPR staging (also used by host pass) ------
  for (int kb = 0; kb < nk; ++kb) {
    const int k0 = kb * 32;
    {
      int idx = t * 16;
      int r = idx >> 5, c = idx & 31;
      const __bf16* g = &A[(size_t)(m0 + r) * K + k0 + c];
      *(bf16x8*)&sA[0][idx]     = *(const bf16x8*)g;
      *(bf16x8*)&sA[0][idx + 8] = *(const bf16x8*)(g + 8);
    }
    {
      int idx = t * 8;
      int r = idx >> 5, c = idx & 31;
      *(bf16x8*)&sB[0][idx] = *(const bf16x8*)&Bt[(size_t)(n0 + r) * K + k0 + c];
    }
    __syncthreads();
    const __bf16* ap = &sA[0][(wave * 16 + l16) * 32 + lhalf * 8];
    v16bf afrag = ld16(ap, ap + 16);
#pragma unroll
    for (int nt = 0; nt < 4; ++nt) {
      const __bf16* bp = &sB[0][(nt * 16 + l16) * 32 + lhalf * 8];
      acc[nt] = wmma_bf16(afrag, ld16(bp, bp + 16), acc[nt]);
    }
    __syncthreads();
  }
#endif

  // epilogue: C layout — VGPR g: rows g (lanes 0-15) / 8+g (lanes 16-31), N=l16
  const int mb = m0 + wave * 16 + lhalf * 8;
#pragma unroll
  for (int nt = 0; nt < 4; ++nt) {
    int col = n0 + nt * 16 + l16;
#pragma unroll
    for (int g = 0; g < 8; ++g) {
      float v = acc[nt][g];
      size_t o = (size_t)(mb + g) * N + col;
      if (Cf) Cf[o] = v + (bias ? bias[col] : 0.f);
      else    Cbf[o] = (__bf16)v;
    }
  }
}

// ---------------------------------------------------------------------------
// 4) RoPE + reshape: qkv[row][3*1024] bf16 ->
//    q,k: [b,h,n,d] bf16 (roped) ; v: [b,h,d,n] bf16 (transposed)
// ---------------------------------------------------------------------------
__global__ __launch_bounds__(256) void rope_kernel(const __bf16* __restrict__ qkv,
                                                   __bf16* __restrict__ q,
                                                   __bf16* __restrict__ k,
                                                   __bf16* __restrict__ vT) {
  unsigned idx = blockIdx.x * 256 + threadIdx.x;   // over BATCH*SEQ*HEADS*32
  int i = idx & 31; idx >>= 5;                     // rotary pair index (d/2)
  int h = idx & 15; idx >>= 4;
  int n = idx & (SEQ - 1); idx >>= 11;
  int b = idx;                                     // 0..3
  size_t rb = ((size_t)b * SEQ + n) * (3 * INNER);
  int col = h * DHEAD + 2 * i;
  float qe = (float)qkv[rb + col],             qo = (float)qkv[rb + col + 1];
  float ke = (float)qkv[rb + INNER + col],     ko = (float)qkv[rb + INNER + col + 1];
  float ve = (float)qkv[rb + 2 * INNER + col], vo = (float)qkv[rb + 2 * INNER + col + 1];
  // inv_freq = theta^(-2i/64) = exp(-ln(1e4)/32 * i)
  float ang = (float)n * __expf(-0.28782313662425572f * (float)i);
  float sn, cs;
  __sincosf(ang, &sn, &cs);
  size_t qkb = (((size_t)(b * HEADS + h)) * SEQ + n) * DHEAD + 2 * i;
  q[qkb]     = (__bf16)(qe * cs - qo * sn);
  q[qkb + 1] = (__bf16)(qe * sn + qo * cs);
  k[qkb]     = (__bf16)(ke * cs - ko * sn);
  k[qkb + 1] = (__bf16)(ke * sn + ko * cs);
  size_t vb = (((size_t)(b * HEADS + h)) * DHEAD + 2 * i) * SEQ + n;
  vT[vb]        = (__bf16)ve;
  vT[vb + SEQ]  = (__bf16)vo;
}

// ---------------------------------------------------------------------------
// 5) Flash attention: grid (SEQ/64, b*h), 128 thr = 4 waves, wave = 16 queries
//    QK^T and P*V via bf16 WMMA; online softmax; P staged via per-wave LDS.
// ---------------------------------------------------------------------------
__global__ __launch_bounds__(128) void flash_attn(const __bf16* __restrict__ q,
                                                  const __bf16* __restrict__ k,
                                                  const __bf16* __restrict__ vT,
                                                  __bf16* __restrict__ out) {
  __shared__ __align__(16) __bf16 sP[4][16 * 32];
  const int t = threadIdx.x;
  const int wave = t >> 5, lane = t & 31;
  const int l16 = lane & 15, lhalf = (lane >> 4) & 1;
  const int bh = blockIdx.y;
  const int b = bh >> 4, h = bh & 15;
  const __bf16* qb = q  + (size_t)bh * SEQ * DHEAD;
  const __bf16* kb = k  + (size_t)bh * SEQ * DHEAD;
  const __bf16* vb = vT + (size_t)bh * DHEAD * SEQ;
  const int q0 = blockIdx.x * 64 + wave * 16;

  // Q A-fragments for d chunks [0,32) and [32,64)
  v16bf aq[2];
  {
    const __bf16* qrow = qb + (size_t)(q0 + l16) * DHEAD;
#pragma unroll
    for (int c = 0; c < 2; ++c) {
      const __bf16* p = qrow + c * 32 + lhalf * 8;
      aq[c] = ld16(p, p + 16);
    }
  }

  v8f o[4];
#pragma unroll
  for (int i = 0; i < 4; ++i) o[i] = v8f_zero();
  float m[8], l[8];
#pragma unroll
  for (int g = 0; g < 8; ++g) { m[g] = -3.0e38f; l[g] = 0.f; }

  __bf16* Pw = sP[wave];

  for (int j0 = 0; j0 < SEQ; j0 += 32) {
    // prefetch next key/value block (global_prefetch_b8)
    if (j0 + 32 < SEQ) {
      __builtin_prefetch(kb + (size_t)(j0 + 32 + l16) * DHEAD, 0, 0);
      __builtin_prefetch(vb + (size_t)l16 * SEQ + j0 + 32, 0, 0);
    }

    // ---- S = Q * K^T, two 16x16 key tiles -------------------------------
    v8f s0 = v8f_zero(), s1 = v8f_zero();
#pragma unroll
    for (int c = 0; c < 2; ++c) {
      const __bf16* k0p = kb + (size_t)(j0 + l16) * DHEAD + c * 32 + lhalf * 8;
      s0 = wmma_bf16(aq[c], ld16(k0p, k0p + 16), s0);
      const __bf16* k1p = kb + (size_t)(j0 + 16 + l16) * DHEAD + c * 32 + lhalf * 8;
      s1 = wmma_bf16(aq[c], ld16(k1p, k1p + 16), s1);
    }

    // ---- online softmax ------------------------------------------------
    float rmax[8];
#pragma unroll
    for (int g = 0; g < 8; ++g) {
      s0[g] = s0[g] * SCALE;
      s1[g] = s1[g] * SCALE;
      rmax[g] = fmaxf(s0[g], s1[g]);
    }
#pragma unroll
    for (int ms = 1; ms < 16; ms <<= 1)
#pragma unroll
      for (int g = 0; g < 8; ++g)
        rmax[g] = fmaxf(rmax[g], __shfl_xor(rmax[g], ms, 32));

    float p0a[8], p1a[8], rsum[8], alpha[8];
#pragma unroll
    for (int g = 0; g < 8; ++g) {
      float mn = fmaxf(m[g], rmax[g]);
      alpha[g] = __expf(m[g] - mn);
      m[g] = mn;
      p0a[g] = __expf(s0[g] - mn);
      p1a[g] = __expf(s1[g] - mn);
      rsum[g] = p0a[g] + p1a[g];
    }
#pragma unroll
    for (int ms = 1; ms < 16; ms <<= 1)
#pragma unroll
      for (int g = 0; g < 8; ++g)
        rsum[g] += __shfl_xor(rsum[g], ms, 32);
#pragma unroll
    for (int g = 0; g < 8; ++g) l[g] = l[g] * alpha[g] + rsum[g];
#pragma unroll
    for (int nt = 0; nt < 4; ++nt)
#pragma unroll
      for (int g = 0; g < 8; ++g) o[nt][g] = o[nt][g] * alpha[g];

    // ---- stage P (C layout -> A layout) via per-wave LDS ---------------
#pragma unroll
    for (int g = 0; g < 8; ++g) {
      Pw[(g + lhalf * 8) * 32 + l16]      = (__bf16)p0a[g];
      Pw[(g + lhalf * 8) * 32 + 16 + l16] = (__bf16)p1a[g];
    }
    asm volatile("s_wait_dscnt 0" ::: "memory");  // DS in-order per wave; explicit
    const __bf16* pr = Pw + l16 * 32 + lhalf * 8;
    v16bf ap = ld16(pr, pr + 16);

    // ---- O += P * V (V transposed: [d][n], keys contiguous) -------------
#pragma unroll
    for (int nt = 0; nt < 4; ++nt) {
      const __bf16* vp = vb + (size_t)(nt * 16 + l16) * SEQ + j0 + lhalf * 8;
      o[nt] = wmma_bf16(ap, ld16(vp, vp + 16), o[nt]);
    }
  }

  // ---- finalize + store to [b][n][h*64+d] bf16 --------------------------
#pragma unroll
  for (int g = 0; g < 8; ++g) {
    size_t row = (size_t)(b * SEQ + q0 + g + lhalf * 8) * INNER + h * DHEAD;
    float inv = 1.f / l[g];
#pragma unroll
    for (int nt = 0; nt < 4; ++nt)
      out[row + nt * 16 + l16] = (__bf16)(o[nt][g] * inv);
  }
}

// ---------------------------------------------------------------------------
// Launcher
// ---------------------------------------------------------------------------
extern "C" void kernel_launch(void* const* d_in, const int* in_sizes, int n_in,
                              void* d_out, int out_size, void* d_ws, size_t ws_size,
                              hipStream_t stream) {
  const float* x     = (const float*)d_in[0];
  const float* gamma = (const float*)d_in[1];
  const float* beta  = (const float*)d_in[2];
  const float* wqkv  = (const float*)d_in[3];
  const float* wout  = (const float*)d_in[4];
  const float* bout  = (const float*)d_in[5];
  float* out = (float*)d_out;

  char* ws = (char*)d_ws;
  size_t off = 0;
  auto carve = [&](size_t bytes) -> char* {
    char* p = ws + off;
    off += (bytes + 255) & ~(size_t)255;
    return p;
  };
  __bf16* xn    = (__bf16*)carve((size_t)ROWS * DIM * 2);            // 16 MB
  __bf16* wqkvT = (__bf16*)carve((size_t)3 * INNER * DIM * 2);       //  6 MB
  __bf16* woutT = (__bf16*)carve((size_t)DIM * INNER * 2);           //  2 MB
  __bf16* qkvb  = (__bf16*)carve((size_t)ROWS * 3 * INNER * 2);      // 48 MB
  __bf16* qr    = (__bf16*)carve((size_t)BATCH * HEADS * SEQ * DHEAD * 2); // 16 MB
  __bf16* kr    = (__bf16*)carve((size_t)BATCH * HEADS * SEQ * DHEAD * 2); // 16 MB
  __bf16* vt    = (__bf16*)carve((size_t)BATCH * HEADS * SEQ * DHEAD * 2); // 16 MB
  __bf16* ao    = (__bf16*)carve((size_t)ROWS * INNER * 2);          // 16 MB

  ln_kernel<<<ROWS, 256, 0, stream>>>(x, gamma, beta, xn);
  transpose_kernel<<<(DIM * 3 * INNER) / 256, 256, 0, stream>>>(wqkv, wqkvT, DIM, 3 * INNER);
  transpose_kernel<<<(INNER * DIM) / 256, 256, 0, stream>>>(wout, woutT, INNER, DIM);

  gemm_wmma_bf16<<<dim3(ROWS / 128, (3 * INNER) / 64), 256, 0, stream>>>(
      xn, wqkvT, ROWS, 3 * INNER, DIM, qkvb, nullptr, nullptr);

  rope_kernel<<<(BATCH * SEQ * HEADS * (DHEAD / 2)) / 256, 256, 0, stream>>>(
      qkvb, qr, kr, vt);

  flash_attn<<<dim3(SEQ / 64, BATCH * HEADS), 128, 0, stream>>>(qr, kr, vt, ao);

  gemm_wmma_bf16<<<dim3(ROWS / 128, DIM / 64), 256, 0, stream>>>(
      ao, woutT, ROWS, DIM, INNER, nullptr, out, bout);
}
